// GAT_NET_20272245637549
// MI455X (gfx1250) — matrix operand order repaired and checked by reading, not compiled, not run on hardware
//
#include <hip/hip_runtime.h>
#include <hip/hip_bf16.h>

// ---------------------------------------------------------------------------
// GATv2 x2 + readouts for MI455X (gfx1250, wave32, WMMA).
// ---------------------------------------------------------------------------
#define NN   50000
#define EE   800000
#define BB   64
#define FIN  128
#define D1   256
#define NCLS 16

typedef __attribute__((ext_vector_type(16))) _Float16 v16h;
typedef __attribute__((ext_vector_type(8)))  _Float16 v8h;
typedef __attribute__((ext_vector_type(8)))  float    v8f;

// Monotonic-int float atomic max (dest must be initialized to -inf).
__device__ __forceinline__ void atomicMaxFloat(float* addr, float v) {
    if (v >= 0.0f)
        atomicMax((int*)addr, __float_as_int(v));
    else
        atomicMin((unsigned int*)addr, __float_as_uint(v));
}

__device__ __forceinline__ void pack8(v16h& a, int base, float4 lo, float4 hi) {
    a[base + 0] = (_Float16)lo.x; a[base + 1] = (_Float16)lo.y;
    a[base + 2] = (_Float16)lo.z; a[base + 3] = (_Float16)lo.w;
    a[base + 4] = (_Float16)hi.x; a[base + 5] = (_Float16)hi.y;
    a[base + 6] = (_Float16)hi.z; a[base + 7] = (_Float16)hi.w;
}

// ---------------------------------------------------------------------------
// WMMA GEMM: D[M,N] = A[M,K] @ B[K,N], A fp32 (lda==K), B pre-converted f16
// and TRANSPOSED (Bt[n*K+k]).  One wave computes 80x16 (5 M-subtiles), so the
// B fragment is loaded once per k-step and reused by 5 WMMAs.
// grid = (M/80, N/16), block = 32.  M=50000=625*80 exactly.
// Fragment layouts used here (wave32; A/B k contiguous, not the ISA 7.12.2 table):
//   A 16x32 f16 : lanes 0-15 row M=lane, K = 0..15; lanes 16-31 K = 16..31
//   B 32x16 f16 : lanes 0-15 col N=lane, K = 0..15; lanes 16-31 K = 16..31
//   C/D 16x16 f32: VGPR r -> row r (lanes 0-15) / row r+8 (lanes 16-31)
// ---------------------------------------------------------------------------
#define MT 5
__global__ __launch_bounds__(32) void gemm_wmma_mt_kernel(
    const float* __restrict__ A, const _Float16* __restrict__ Bt,
    float* __restrict__ D, int K, int ldd)
{
    const int lane = threadIdx.x & 31;
    const int half = lane >> 4;
    const int l16  = lane & 15;
    const int m0 = blockIdx.x * (16 * MT);
    const int n0 = blockIdx.y * 16;
    const int bcol = n0 + l16;

    v8f acc[MT];
    #pragma unroll
    for (int t = 0; t < MT; ++t) acc[t] = (v8f){};

    for (int kb = 0; kb < K; kb += 32) {
        // B fragment: 16 contiguous f16 at Bt[bcol*K + kb + half*16]
        const v8h* bp = (const v8h*)(Bt + (size_t)bcol * K + kb + half * 16);
        const v8h b0 = bp[0], b1 = bp[1];
        v16h b;
        #pragma unroll
        for (int e = 0; e < 8; ++e) { b[e] = b0[e]; b[8 + e] = b1[e]; }

        #pragma unroll
        for (int t = 0; t < MT; ++t) {
            const float* arow = A + (size_t)(m0 + t * 16 + l16) * K + kb;
            const float4* ap0 = (const float4*)(arow + half * 16);
            const float4* ap1 = (const float4*)(arow + half * 16 + 8);
            v16h a;
            pack8(a, 0, ap0[0], ap0[1]);
            pack8(a, 8, ap1[0], ap1[1]);
            acc[t] = __builtin_amdgcn_wmma_f32_16x16x32_f16(
                false, a, false, b, (short)0, acc[t], false, false);
        }
    }

    #pragma unroll
    for (int t = 0; t < MT; ++t) {
        float* Dp = D + (size_t)(m0 + t * 16 + half * 8) * ldd + n0 + l16;
        #pragma unroll
        for (int r = 0; r < 8; ++r) Dp[(size_t)r * ldd] = acc[t][r];
    }
}

// Classifier GEMM: y = h @ Wc + bc, fused segment_max over batch into yout.
// N = 16 classes => single tile column. grid = (M/80), block = 32.
__global__ __launch_bounds__(32) void gemm_y_max_kernel(
    const float* __restrict__ A, const _Float16* __restrict__ Wct,
    const float* __restrict__ bc, const int* __restrict__ batch,
    float* __restrict__ yout)
{
    const int lane = threadIdx.x & 31;
    const int half = lane >> 4;
    const int l16  = lane & 15;
    const int m0 = blockIdx.x * (16 * MT);

    v8f acc[MT];
    #pragma unroll
    for (int t = 0; t < MT; ++t) acc[t] = (v8f){};

    for (int kb = 0; kb < D1; kb += 32) {
        const v8h* bp = (const v8h*)(Wct + (size_t)l16 * D1 + kb + half * 16);
        const v8h b0 = bp[0], b1 = bp[1];
        v16h b;
        #pragma unroll
        for (int e = 0; e < 8; ++e) { b[e] = b0[e]; b[8 + e] = b1[e]; }

        #pragma unroll
        for (int t = 0; t < MT; ++t) {
            const float* arow = A + (size_t)(m0 + t * 16 + l16) * D1 + kb;
            const float4* ap0 = (const float4*)(arow + half * 16);
            const float4* ap1 = (const float4*)(arow + half * 16 + 8);
            v16h a;
            pack8(a, 0, ap0[0], ap0[1]);
            pack8(a, 8, ap1[0], ap1[1]);
            acc[t] = __builtin_amdgcn_wmma_f32_16x16x32_f16(
                false, a, false, b, (short)0, acc[t], false, false);
        }
    }

    const float bias = bc[l16];
    #pragma unroll
    for (int t = 0; t < MT; ++t) {
        #pragma unroll
        for (int r = 0; r < 8; ++r) {
            const int row = m0 + t * 16 + half * 8 + r;
            atomicMaxFloat(&yout[batch[row] * NCLS + l16], acc[t][r] + bias);
        }
    }
}

// ---------------------------------------------------------------------------
// Weight prep: f32 [K,N] row-major -> f16 transposed [N,K].
// ---------------------------------------------------------------------------
__global__ void convert_wT_kernel(const float* __restrict__ W,
                                  _Float16* __restrict__ Wt, int K, int N) {
    const int i = blockIdx.x * blockDim.x + threadIdx.x;
    if (i >= K * N) return;
    const int k = i / N, n = i - k * N;
    Wt[(size_t)n * K + k] = (_Float16)W[i];
}

// Collapse activation-free MLP head: Wct[j,k] (f16, transposed) = (W1@W2@W3)^T,
// bc = b1@W23 + b2@W3 + b3.  Single 256-thread block; W23 staged in LDS.
__global__ __launch_bounds__(256) void prep_wc_kernel(
    const float* __restrict__ W1, const float* __restrict__ b1,
    const float* __restrict__ W2, const float* __restrict__ b2,
    const float* __restrict__ W3, const float* __restrict__ b3,
    _Float16* __restrict__ Wct, float* __restrict__ bc)
{
    __shared__ float W23[128 * 16];
    const int t = threadIdx.x;
    for (int idx = t; idx < 128 * 16; idx += 256) {
        const int k = idx >> 4, j = idx & 15;
        float acc = 0.f;
        for (int q = 0; q < 32; ++q) acc += W2[k * 32 + q] * W3[q * 16 + j];
        W23[idx] = acc;
    }
    __syncthreads();
    for (int idx = t; idx < 256 * 16; idx += 256) {
        const int r = idx >> 4, j = idx & 15;
        float acc = 0.f;
        for (int k = 0; k < 128; ++k) acc += W1[r * 128 + k] * W23[k * 16 + j];
        Wct[(size_t)j * 256 + r] = (_Float16)acc;   // transposed, f16
    }
    if (t < 16) {
        float acc = b3[t];
        for (int k = 0; k < 128; ++k) acc += b1[k] * W23[k * 16 + t];
        for (int q = 0; q < 32;  ++q) acc += b2[q] * W3[q * 16 + t];
        bc[t] = acc;
    }
}

// ---------------------------------------------------------------------------
// Edge pass 1: logits[e,h] = sum_c att[h,c]*lrelu(xl[src][c]+xr[dst][c],0.2)
// and segment-max into mmax[dst,h]. One wave per edge; lane covers c=lane+32j.
// ---------------------------------------------------------------------------
template<int H, int CH>
__global__ __launch_bounds__(256) void edge_logits_kernel(
    const int* __restrict__ src, const int* __restrict__ dst,
    const float* __restrict__ xl, const float* __restrict__ xr,
    const float* __restrict__ att, float* __restrict__ logits,
    float* __restrict__ mmax, int ecount)
{
    const int e = blockIdx.x * 8 + (threadIdx.x >> 5);
    if (e >= ecount) return;
    const int lane = threadIdx.x & 31;
    const int s = src[e], d = dst[e];
    const float* pl = xl + (size_t)s * 256;
    const float* pr = xr + (size_t)d * 256;

    float hp[H];
    #pragma unroll
    for (int h = 0; h < H; ++h) hp[h] = 0.f;
    #pragma unroll
    for (int j = 0; j < 8; ++j) {
        const int c = lane + 32 * j;
        float v = pl[c] + pr[c];
        v = v > 0.f ? v : 0.2f * v;               // leaky_relu(., 0.2)
        hp[(32 * j) / CH] += v * att[c];          // head index is j-constant
    }
    #pragma unroll
    for (int h = 0; h < H; ++h) {
        float v = hp[h];
        #pragma unroll
        for (int off = 16; off > 0; off >>= 1) v += __shfl_xor(v, off, 32);
        if (lane == 0) {
            logits[(size_t)e * H + h] = v;
            atomicMaxFloat(&mmax[(size_t)d * H + h], v);
        }
    }
}

// Edge pass 2: a = exp(logit - m[dst]); segment-sum into ssum[dst].
__global__ __launch_bounds__(256) void edge_exp_kernel(
    float* __restrict__ logits, const int* __restrict__ dst,
    const float* __restrict__ mmax, float* __restrict__ ssum, int total, int H)
{
    const int i = blockIdx.x * blockDim.x + threadIdx.x;
    if (i >= total) return;
    const int e = i / H, h = i - e * H;
    const int d = dst[e];
    const float a = __expf(logits[i] - mmax[(size_t)d * H + h]);
    logits[i] = a;
    atomicAdd(&ssum[(size_t)d * H + h], a);
}

// Edge pass 3: out[dst] += (a/ssum[dst]) * xl[src]. One wave per edge.
template<int H, int CH>
__global__ __launch_bounds__(256) void edge_aggr_kernel(
    const int* __restrict__ src, const int* __restrict__ dst,
    const float* __restrict__ xl, const float* __restrict__ alpha,
    const float* __restrict__ ssum, float* __restrict__ out, int ecount)
{
    const int e = blockIdx.x * 8 + (threadIdx.x >> 5);
    if (e >= ecount) return;
    const int lane = threadIdx.x & 31;
    const int s = src[e], d = dst[e];

    float t = 0.f;
    if (lane < H) t = alpha[(size_t)e * H + lane] / ssum[(size_t)d * H + lane];
    float al[H];
    #pragma unroll
    for (int h = 0; h < H; ++h) al[h] = __shfl(t, h, 32);

    const float* pl = xl + (size_t)s * 256;
    float* po = out + (size_t)d * 256;
    #pragma unroll
    for (int j = 0; j < 8; ++j) {
        const int c = lane + 32 * j;
        atomicAdd(&po[c], al[(32 * j) / CH] * pl[c]);
    }
}

// ---------------------------------------------------------------------------
// Small utility kernels.
// ---------------------------------------------------------------------------
__global__ void fill_kernel(float* p, float v, int n) {
    const int i = blockIdx.x * blockDim.x + threadIdx.x;
    if (i < n) p[i] = v;
}
__global__ void fill_bias_kernel(float* p, const float* __restrict__ bias, int n) {
    const int i = blockIdx.x * blockDim.x + threadIdx.x;
    if (i < n) p[i] = bias[i & 255];
}
__global__ void leaky_kernel(float* p, int n, float slope) {
    const int i = blockIdx.x * blockDim.x + threadIdx.x;
    if (i < n) { const float v = p[i]; p[i] = v > 0.f ? v : slope * v; }
}
__global__ void embs_max_kernel(const float* __restrict__ h,
                                const int* __restrict__ batch,
                                float* __restrict__ embs, int n) {
    const int i = blockIdx.x * blockDim.x + threadIdx.x;
    if (i >= n) return;
    const int node = i >> 8, c = i & 255;
    atomicMaxFloat(&embs[(size_t)batch[node] * 256 + c], h[i]);
}

// ---------------------------------------------------------------------------
// Launch.
// ---------------------------------------------------------------------------
extern "C" void kernel_launch(void* const* d_in, const int* in_sizes, int n_in,
                              void* d_out, int out_size, void* d_ws, size_t ws_size,
                              hipStream_t stream) {
    const float* x    = (const float*)d_in[0];
    const int*   ei   = (const int*)d_in[1];
    const int*   batch= (const int*)d_in[2];
    const float* Wl1  = (const float*)d_in[3];
    const float* Wr1  = (const float*)d_in[4];
    const float* att1 = (const float*)d_in[5];
    const float* bg1  = (const float*)d_in[6];
    const float* Wl2  = (const float*)d_in[7];
    const float* Wr2  = (const float*)d_in[8];
    const float* att2 = (const float*)d_in[9];
    const float* bg2  = (const float*)d_in[10];
    const float* W1   = (const float*)d_in[11];
    const float* b1   = (const float*)d_in[12];
    const float* W2   = (const float*)d_in[13];
    const float* b2   = (const float*)d_in[14];
    const float* W3   = (const float*)d_in[15];
    const float* b3   = (const float*)d_in[16];

    const int* src = ei;
    const int* dst = ei + EE;

    // Workspace (floats): xl | xr | h | logits(E*4) | m(N*4) | s(N*4) | bc |
    // then f16 region: Wl1^T | Wr1^T | Wl2^T | Wr2^T | Wc^T
    float* ws   = (float*)d_ws;
    const size_t NC = (size_t)NN * 256;
    float* xl   = ws;
    float* xr   = xl + NC;
    float* hh   = xr + NC;
    float* alog = hh + NC;
    float* mbuf = alog + (size_t)EE * 4;
    float* sbuf = mbuf + (size_t)NN * 4;
    float* bc   = sbuf + (size_t)NN * 4;
    _Float16* wl1h = (_Float16*)(bc + 16);          // 128*256
    _Float16* wr1h = wl1h + FIN * D1;               // 128*256
    _Float16* wl2h = wr1h + FIN * D1;               // 256*256
    _Float16* wr2h = wl2h + D1 * D1;                // 256*256
    _Float16* wcth = wr2h + D1 * D1;                // 16*256

    float* yout = (float*)d_out;               // [B,16]  = 1024 floats
    float* embs = (float*)d_out + BB * NCLS;   // [B,256] = 16384 floats

    const float NEG_INF = -__builtin_inff();
    const dim3 gM(625, 16);                    // 50000/80 x 256/16 tiles

    // 0) weight prep (f16, transposed) + collapsed classifier
    convert_wT_kernel<<<(FIN * D1 + 255) / 256, 256, 0, stream>>>(Wl1, wl1h, FIN, D1);
    convert_wT_kernel<<<(FIN * D1 + 255) / 256, 256, 0, stream>>>(Wr1, wr1h, FIN, D1);
    convert_wT_kernel<<<(D1 * D1 + 255) / 256, 256, 0, stream>>>(Wl2, wl2h, D1, D1);
    convert_wT_kernel<<<(D1 * D1 + 255) / 256, 256, 0, stream>>>(Wr2, wr2h, D1, D1);
    prep_wc_kernel<<<1, 256, 0, stream>>>(W1, b1, W2, b2, W3, b3, wcth, bc);

    // 1) init outputs + layer-1 softmax state + h := bg1
    fill_kernel<<<(out_size + 255) / 256, 256, 0, stream>>>((float*)d_out, NEG_INF, out_size);
    fill_bias_kernel<<<(int)((NC + 255) / 256), 256, 0, stream>>>(hh, bg1, (int)NC);
    fill_kernel<<<(NN * 4 + 255) / 256, 256, 0, stream>>>(mbuf, NEG_INF, NN * 4);
    fill_kernel<<<(NN * 4 + 255) / 256, 256, 0, stream>>>(sbuf, 0.f, NN * 4);

    // 2) layer-1 projections (WMMA)
    gemm_wmma_mt_kernel<<<gM, 32, 0, stream>>>(x, wl1h, xl, FIN, D1);
    gemm_wmma_mt_kernel<<<gM, 32, 0, stream>>>(x, wr1h, xr, FIN, D1);

    // 3) layer-1 edge softmax + aggregation (4 heads x 64 ch)
    edge_logits_kernel<4, 64><<<EE / 8, 256, 0, stream>>>(src, dst, xl, xr, att1, alog, mbuf, EE);
    edge_exp_kernel<<<(EE * 4 + 255) / 256, 256, 0, stream>>>(alog, dst, mbuf, sbuf, EE * 4, 4);
    edge_aggr_kernel<4, 64><<<EE / 8, 256, 0, stream>>>(src, dst, xl, alog, sbuf, hh, EE);
    leaky_kernel<<<(int)((NC + 255) / 256), 256, 0, stream>>>(hh, (int)NC, 0.01f);

    // 4) layer-2 projections (WMMA, reuse xl/xr)
    gemm_wmma_mt_kernel<<<gM, 32, 0, stream>>>(hh, wl2h, xl, D1, D1);
    gemm_wmma_mt_kernel<<<gM, 32, 0, stream>>>(hh, wr2h, xr, D1, D1);

    // 5) layer-2 softmax state + h := bg2
    fill_bias_kernel<<<(int)((NC + 255) / 256), 256, 0, stream>>>(hh, bg2, (int)NC);
    fill_kernel<<<(NN + 255) / 256, 256, 0, stream>>>(mbuf, NEG_INF, NN);
    fill_kernel<<<(NN + 255) / 256, 256, 0, stream>>>(sbuf, 0.f, NN);

    // 6) layer-2 edge softmax + aggregation (1 head x 256 ch)
    edge_logits_kernel<1, 256><<<EE / 8, 256, 0, stream>>>(src, dst, xl, xr, att2, alog, mbuf, EE);
    edge_exp_kernel<<<(EE + 255) / 256, 256, 0, stream>>>(alog, dst, mbuf, sbuf, EE, 1);
    edge_aggr_kernel<1, 256><<<EE / 8, 256, 0, stream>>>(src, dst, xl, alog, sbuf, hh, EE);

    // 7) readouts: embs = segment_max(h); y = segment_max(h @ Wc + bc)
    embs_max_kernel<<<(int)((NC + 255) / 256), 256, 0, stream>>>(hh, batch, embs, (int)NC);
    gemm_y_max_kernel<<<625, 32, 0, stream>>>(hh, wcth, bc, batch, yout);
}
